// GAT_86844238725804
// MI455X (gfx1250) — matrix operand order repaired and checked by reading, not compiled, not run on hardware
//
#include <hip/hip_runtime.h>

typedef __attribute__((ext_vector_type(16))) _Float16 v16h;
typedef __attribute__((ext_vector_type(8)))  float    v8f;

constexpr int   B = 2, N = 4096, D = 8, H = 4, HID = 32, EN = 8;
constexpr float ALPHA   = 0.2f;
constexpr float NEG_INF = -9e15f;

// ---- async global->LDS path (gfx1250), guarded so the file always compiles
#if defined(__gfx1250__) &&                                          \
    __has_builtin(__builtin_amdgcn_global_load_async_to_lds_b128) && \
    __has_builtin(__builtin_amdgcn_global_load_async_to_lds_b32) &&  \
    __has_builtin(__builtin_amdgcn_s_wait_asynccnt)
#define GAT_ASYNC 1
#else
#define GAT_ASYNC 0
#endif

#if GAT_ASYNC
// Builtin signatures (from clang diagnostics): b128 takes int-vector(4)
// pointers, b32 takes int pointers; arg0 in AS(1) (global), arg1 in AS(3)
// (LDS), then imm offset + imm cpol.
typedef int gat_v4i __attribute__((vector_size(16)));
typedef __attribute__((address_space(1))) gat_v4i* gp_b128;
typedef __attribute__((address_space(3))) gat_v4i* lp_b128;
typedef __attribute__((address_space(1))) int*     gp_b32;
typedef __attribute__((address_space(3))) int*     lp_b32;
// flat address == global address numerically; LDS offset == low 32 bits of the
// flat LDS-aperture address (aperture check uses ADDR[63:32], ISA 10.2).
__device__ __forceinline__ void async_b128(const void* g, void* l) {
  __builtin_amdgcn_global_load_async_to_lds_b128(
      (gp_b128)(unsigned long long)g,
      (lp_b128)(unsigned int)(unsigned long long)l, 0, 0);
}
__device__ __forceinline__ void async_b32(const void* g, void* l) {
  __builtin_amdgcn_global_load_async_to_lds_b32(
      (gp_b32)(unsigned long long)g,
      (lp_b32)(unsigned int)(unsigned long long)l, 0, 0);
}
#endif

// ---------------------------------------------------------------------------
// K1: layer-1 projection. wh = x @ W[h] (f16 copy for WMMA B operand),
//     plus attention scalars wh1 = wh·a_top, wh2 = wh·a_bot.
// ---------------------------------------------------------------------------
__global__ __launch_bounds__(256) void gat1_project(
    const float* __restrict__ x, const float* __restrict__ W,
    const float* __restrict__ a, _Float16* __restrict__ wh_h,
    float* __restrict__ s1a, float* __restrict__ s1b) {
  __shared__ float sW[D * HID];
  __shared__ float sa[2 * HID];
  int t  = blockIdx.x * 256 + threadIdx.x;
  int bh = t / N;            // b*H + h  (constant per block: N % 256 == 0)
  int n  = t % N;
  int h  = bh % H;
  if (threadIdx.x < D * HID) sW[threadIdx.x] = W[h * D * HID + threadIdx.x];
  if (threadIdx.x < 2 * HID) sa[threadIdx.x] = a[h * 2 * HID + threadIdx.x];
  __syncthreads();
  float xv[D];
  const float* xp = x + ((size_t)(bh / H) * N + n) * D;
#pragma unroll
  for (int d = 0; d < D; ++d) xv[d] = xp[d];
  float acc1 = 0.f, acc2 = 0.f;
  _Float16* whp = wh_h + ((size_t)bh * N + n) * HID;
#pragma unroll
  for (int c = 0; c < HID; ++c) {
    float w = 0.f;
#pragma unroll
    for (int d = 0; d < D; ++d) w += xv[d] * sW[d * HID + c];
    whp[c] = (_Float16)w;
    acc1 += w * sa[c];
    acc2 += w * sa[HID + c];
  }
  s1a[bh * N + n] = acc1;
  s1b[bh * N + n] = acc2;
}

// ---------------------------------------------------------------------------
// K2/K5: per-row softmax stats (max, sum-exp) over masked scores.
// One wave32 per row; adj row (16 KB) is L2-resident on the 2nd pass.
// ---------------------------------------------------------------------------
__global__ __launch_bounds__(256) void gat_softmax_stats(
    const float* __restrict__ adj, const float* __restrict__ sa_,
    const float* __restrict__ sb_, float* __restrict__ m_out,
    float* __restrict__ s_out, int heads) {
  int wave = blockIdx.x * 8 + (threadIdx.x >> 5);
  int lane = threadIdx.x & 31;
  int b = wave / (heads * N);
  int i = wave % N;
  float w1 = sa_[wave];
  const float* arow = adj + (size_t)b * N * N + (size_t)i * N;
  const float* sbb  = sb_ + (size_t)(wave / N) * N;
  float m = NEG_INF;
  for (int j = lane; j < N; j += 32) {
    float av = arow[j];
    float e  = w1 + sbb[j];
    e = e > 0.f ? e : ALPHA * e;
    float sc = (av != 0.f) ? e * av : NEG_INF;
    m = fmaxf(m, sc);
  }
#pragma unroll
  for (int off = 16; off > 0; off >>= 1) m = fmaxf(m, __shfl_xor(m, off));
  float s = 0.f;
  for (int j = lane; j < N; j += 32) {
    float av = arow[j];
    float e  = w1 + sbb[j];
    e = e > 0.f ? e : ALPHA * e;
    float sc = (av != 0.f) ? e * av : NEG_INF;
    s += __expf(sc - m);
  }
#pragma unroll
  for (int off = 16; off > 0; off >>= 1) s += __shfl_xor(s, off);
  if (lane == 0) { m_out[wave] = m; s_out[wave] = s; }
}

// ---------------------------------------------------------------------------
// K3: layer-1 att @ wh via v_wmma_f32_16x16x32_f16, fused ELU + head-concat.
// One wave per (b,h, 16-row tile). Async double-buffered LDS staging when the
// GLOBAL_LOAD_ASYNC_TO_LDS builtins are available (9 async ops / tile,
// s_wait_asynccnt 9 pipelines tile t+1's DMA under tile t's exp+WMMA).
// ---------------------------------------------------------------------------
__global__ __launch_bounds__(32) void gat1_attn_wmma(
    const float* __restrict__ adj, const _Float16* __restrict__ wh_h,
    const float* __restrict__ s1a, const float* __restrict__ s1b,
    const float* __restrict__ m1, const float* __restrict__ s1,
    float* __restrict__ hcat) {
  int tile = blockIdx.x;
  int iblk = tile % (N / 16);
  int bh   = tile / (N / 16);
  int b = bh / H, h = bh % H;
  int i0   = iblk * 16;
  int lane = threadIdx.x;
  int hi   = lane >> 4;        // lane-half selects K sub-range
  int lrow = lane & 15;        // A row / B col for this lane
  int myrow = i0 + lrow;
  float mrow  = m1[bh * N + myrow];
  float inv_s = 1.f / s1[bh * N + myrow];
  float w1    = s1a[bh * N + myrow];
  const float* adj_b = adj + (size_t)b * N * N;
  v8f c0 = {}, c1 = {};

  auto compute = [&](const float* La, const float* Lw2, const _Float16* Lwh) {
    v16h aop, b0v, b1v;
#pragma unroll
    for (int e = 0; e < 16; ++e) {
      int k = e + (hi ? 16 : 0);
      float av = La[lrow * 32 + k];
      float ew = w1 + Lw2[k];
      ew = ew > 0.f ? ew : ALPHA * ew;
      float sc = (av != 0.f) ? ew * av : NEG_INF;
      aop[e] = (_Float16)(__expf(sc - mrow) * inv_s);
    }
    int kb = hi ? 16 : 0;
#pragma unroll
    for (int e = 0; e < 16; ++e) {
      b0v[e] = Lwh[(kb + e) * HID + lrow];
      b1v[e] = Lwh[(kb + e) * HID + 16 + lrow];
    }
    c0 = __builtin_amdgcn_wmma_f32_16x16x32_f16(false, aop, false, b0v,
                                                (short)0, c0, false, false);
    c1 = __builtin_amdgcn_wmma_f32_16x16x32_f16(false, aop, false, b1v,
                                                (short)0, c1, false, false);
  };

#if GAT_ASYNC
  __shared__ __align__(16) float    lds_adj[2][16 * 32];
  __shared__ __align__(16) float    lds_w2 [2][32];
  __shared__ __align__(16) _Float16 lds_wh [2][32 * HID];
  auto issue = [&](int buf, int j0) {   // 9 async ops per 32-wide tile
    int ar = lane >> 1, ac = (lane & 1) * 16;
    const float* ga = adj_b + (size_t)(i0 + ar) * N + j0 + ac;
    float* la = &lds_adj[buf][ar * 32 + ac];
#pragma unroll
    for (int q = 0; q < 4; ++q) async_b128(ga + 4 * q, la + 4 * q);
    async_b32(s1b + (size_t)bh * N + j0 + lane, &lds_w2[buf][lane]);
    const _Float16* gw = wh_h + ((size_t)bh * N + j0 + lane) * HID;
    _Float16* lw = &lds_wh[buf][lane * HID];
#pragma unroll
    for (int q = 0; q < 4; ++q) async_b128(gw + 8 * q, lw + 8 * q);
  };
  issue(0, 0);
  for (int j0 = 0; j0 < N; j0 += 32) {
    int buf = (j0 >> 5) & 1;
    if (j0 + 32 < N) {
      issue(buf ^ 1, j0 + 32);
      __builtin_amdgcn_s_wait_asynccnt(9);  // current tile resident (in-order)
    } else {
      __builtin_amdgcn_s_wait_asynccnt(0);
    }
    compute(lds_adj[buf], lds_w2[buf], lds_wh[buf]);
  }
#else
  __shared__ __align__(16) float    lds_adj[16 * 32];
  __shared__ __align__(16) float    lds_w2[32];
  __shared__ __align__(16) _Float16 lds_wh[32 * HID];
  for (int j0 = 0; j0 < N; j0 += 32) {
    int ar = lane >> 1, ac = (lane & 1) * 16;
    const float4* asrc = (const float4*)(adj_b + (size_t)(i0 + ar) * N + j0 + ac);
    float4* adst = (float4*)(lds_adj + ar * 32 + ac);
    adst[0] = asrc[0]; adst[1] = asrc[1]; adst[2] = asrc[2]; adst[3] = asrc[3];
    lds_w2[lane] = s1b[bh * N + j0 + lane];
    const uint4* wsrc = (const uint4*)(wh_h + ((size_t)bh * N + j0 + lane) * HID);
    uint4* wdst = (uint4*)(lds_wh + lane * HID);
    wdst[0] = wsrc[0]; wdst[1] = wsrc[1]; wdst[2] = wsrc[2]; wdst[3] = wsrc[3];
    if (j0 + 32 < N)
      __builtin_prefetch(adj_b + (size_t)(i0 + ar) * N + j0 + 32 + ac, 0, 0);
    __syncthreads();
    compute(lds_adj, lds_w2, lds_wh);
    __syncthreads();
  }
#endif
  // C layout: VGPR r -> row r (lanes 0-15) / row 8+r (lanes 16-31), col = lane%16
#pragma unroll
  for (int r = 0; r < 8; ++r) {
    int row = i0 + r + hi * 8;
    float v0 = c0[r]; v0 = v0 > 0.f ? v0 : __expf(v0) - 1.f;  // ELU
    float v1 = c1[r]; v1 = v1 > 0.f ? v1 : __expf(v1) - 1.f;
    size_t base = ((size_t)b * N + row) * (H * HID) + h * HID;
    hcat[base + lrow]      = v0;
    hcat[base + 16 + lrow] = v1;
  }
}

// ---------------------------------------------------------------------------
// K4: layer-2 projection. wh2 = hcat @ W_last (f16 copy) + scalar dots.
// ---------------------------------------------------------------------------
__global__ __launch_bounds__(256) void gat2_project(
    const float* __restrict__ hcat, const float* __restrict__ W_last,
    const float* __restrict__ a_last, _Float16* __restrict__ wh2_h,
    float* __restrict__ s2a, float* __restrict__ s2b) {
  __shared__ float sW[H * HID * EN];  // 128x8
  __shared__ float sa[2 * EN];
  int tid = threadIdx.x;
#pragma unroll
  for (int q = 0; q < 4; ++q) sW[tid + q * 256] = W_last[tid + q * 256];
  if (tid < 2 * EN) sa[tid] = a_last[tid];
  __syncthreads();
  int t = blockIdx.x * 256 + tid;  // b*N + n
  const float* hrow = hcat + (size_t)t * (H * HID);
  float acc[EN] = {};
  for (int d = 0; d < H * HID; ++d) {
    float hv = hrow[d];
#pragma unroll
    for (int c = 0; c < EN; ++c) acc[c] += hv * sW[d * EN + c];
  }
  float d1 = 0.f, d2 = 0.f;
  _Float16* wp = wh2_h + (size_t)t * EN;
#pragma unroll
  for (int c = 0; c < EN; ++c) {
    wp[c] = (_Float16)acc[c];
    d1 += acc[c] * sa[c];
    d2 += acc[c] * sa[EN + c];
  }
  s2a[t] = d1;
  s2b[t] = d2;
}

// ---------------------------------------------------------------------------
// K6: layer-2 att @ wh2 (EN=8, B cols 8..15 zero-padded), fused ELU -> out.
// ---------------------------------------------------------------------------
__global__ __launch_bounds__(32) void gat2_attn_wmma(
    const float* __restrict__ adj, const _Float16* __restrict__ wh2_h,
    const float* __restrict__ s2a, const float* __restrict__ s2b,
    const float* __restrict__ m2, const float* __restrict__ s2,
    float* __restrict__ out) {
  int tile = blockIdx.x;
  int iblk = tile % (N / 16);
  int b    = tile / (N / 16);
  int i0   = iblk * 16;
  int lane = threadIdx.x;
  int hi   = lane >> 4;
  int lrow = lane & 15;
  int myrow = i0 + lrow;
  float mrow  = m2[b * N + myrow];
  float inv_s = 1.f / s2[b * N + myrow];
  float w1    = s2a[b * N + myrow];
  const float* adj_b = adj + (size_t)b * N * N;
  v8f c0 = {};

  auto compute = [&](const float* La, const float* Lw2, const _Float16* Lwh) {
    v16h aop, b0v;
#pragma unroll
    for (int e = 0; e < 16; ++e) {
      int k = e + (hi ? 16 : 0);
      float av = La[lrow * 32 + k];
      float ew = w1 + Lw2[k];
      ew = ew > 0.f ? ew : ALPHA * ew;
      float sc = (av != 0.f) ? ew * av : NEG_INF;
      aop[e] = (_Float16)(__expf(sc - mrow) * inv_s);
    }
    int kb = hi ? 16 : 0;
#pragma unroll
    for (int e = 0; e < 16; ++e)
      b0v[e] = (lrow < EN) ? Lwh[(kb + e) * EN + lrow] : (_Float16)0.f;
    c0 = __builtin_amdgcn_wmma_f32_16x16x32_f16(false, aop, false, b0v,
                                                (short)0, c0, false, false);
  };

#if GAT_ASYNC
  __shared__ __align__(16) float    lds_adj[2][16 * 32];
  __shared__ __align__(16) float    lds_w2 [2][32];
  __shared__ __align__(16) _Float16 lds_wh [2][32 * EN];
  auto issue = [&](int buf, int j0) {   // 6 async ops per tile
    int ar = lane >> 1, ac = (lane & 1) * 16;
    const float* ga = adj_b + (size_t)(i0 + ar) * N + j0 + ac;
    float* la = &lds_adj[buf][ar * 32 + ac];
#pragma unroll
    for (int q = 0; q < 4; ++q) async_b128(ga + 4 * q, la + 4 * q);
    async_b32(s2b + (size_t)b * N + j0 + lane, &lds_w2[buf][lane]);
    async_b128(wh2_h + ((size_t)b * N + j0 + lane) * EN, &lds_wh[buf][lane * EN]);
  };
  issue(0, 0);
  for (int j0 = 0; j0 < N; j0 += 32) {
    int buf = (j0 >> 5) & 1;
    if (j0 + 32 < N) {
      issue(buf ^ 1, j0 + 32);
      __builtin_amdgcn_s_wait_asynccnt(6);
    } else {
      __builtin_amdgcn_s_wait_asynccnt(0);
    }
    compute(lds_adj[buf], lds_w2[buf], lds_wh[buf]);
  }
#else
  __shared__ __align__(16) float    lds_adj[16 * 32];
  __shared__ __align__(16) float    lds_w2[32];
  __shared__ __align__(16) _Float16 lds_wh[32 * EN];
  for (int j0 = 0; j0 < N; j0 += 32) {
    int ar = lane >> 1, ac = (lane & 1) * 16;
    const float4* asrc = (const float4*)(adj_b + (size_t)(i0 + ar) * N + j0 + ac);
    float4* adst = (float4*)(lds_adj + ar * 32 + ac);
    adst[0] = asrc[0]; adst[1] = asrc[1]; adst[2] = asrc[2]; adst[3] = asrc[3];
    lds_w2[lane] = s2b[b * N + j0 + lane];
    const uint4* wsrc = (const uint4*)(wh2_h + ((size_t)b * N + j0 + lane) * EN);
    ((uint4*)(lds_wh + lane * EN))[0] = wsrc[0];
    if (j0 + 32 < N)
      __builtin_prefetch(adj_b + (size_t)(i0 + ar) * N + j0 + 32 + ac, 0, 0);
    __syncthreads();
    compute(lds_adj, lds_w2, lds_wh);
    __syncthreads();
  }
#endif
  if (lrow < EN) {
#pragma unroll
    for (int r = 0; r < 8; ++r) {
      int row = i0 + r + hi * 8;
      float v = c0[r];
      v = v > 0.f ? v : __expf(v) - 1.f;  // ELU
      out[((size_t)b * N + row) * EN + lrow] = v;
    }
  }
}

// ---------------------------------------------------------------------------
extern "C" void kernel_launch(void* const* d_in, const int* in_sizes, int n_in,
                              void* d_out, int out_size, void* d_ws, size_t ws_size,
                              hipStream_t stream) {
  (void)in_sizes; (void)n_in; (void)out_size; (void)ws_size;
  const float* x      = (const float*)d_in[0];
  const float* adj    = (const float*)d_in[1];
  const float* W      = (const float*)d_in[2];
  const float* a      = (const float*)d_in[3];
  const float* W_last = (const float*)d_in[4];
  const float* a_last = (const float*)d_in[5];
  float* out = (float*)d_out;

  char* ws = (char*)d_ws;
  size_t o = 0;
  auto carve = [&](size_t bytes) -> void* {
    void* p = ws + o;
    o += (bytes + 255) & ~(size_t)255;
    return p;
  };
  _Float16* wh_h  = (_Float16*)carve((size_t)B * H * N * HID * sizeof(_Float16));
  float*    s1a   = (float*)carve((size_t)B * H * N * sizeof(float));
  float*    s1b   = (float*)carve((size_t)B * H * N * sizeof(float));
  float*    m1    = (float*)carve((size_t)B * H * N * sizeof(float));
  float*    s1    = (float*)carve((size_t)B * H * N * sizeof(float));
  float*    hcat  = (float*)carve((size_t)B * N * H * HID * sizeof(float));
  _Float16* wh2_h = (_Float16*)carve((size_t)B * N * EN * sizeof(_Float16));
  float*    s2a   = (float*)carve((size_t)B * N * sizeof(float));
  float*    s2b   = (float*)carve((size_t)B * N * sizeof(float));
  float*    m2    = (float*)carve((size_t)B * N * sizeof(float));
  float*    s2    = (float*)carve((size_t)B * N * sizeof(float));

  gat1_project<<<(B * H * N) / 256, 256, 0, stream>>>(x, W, a, wh_h, s1a, s1b);
  gat_softmax_stats<<<(B * H * N) / 8, 256, 0, stream>>>(adj, s1a, s1b, m1, s1, H);
  gat1_attn_wmma<<<B * H * (N / 16), 32, 0, stream>>>(adj, wh_h, s1a, s1b, m1, s1, hcat);
  gat2_project<<<(B * N) / 256, 256, 0, stream>>>(hcat, W_last, a_last, wh2_h, s2a, s2b);
  gat_softmax_stats<<<(B * N) / 8, 256, 0, stream>>>(adj, s2a, s2b, m2, s2, 1);
  gat2_attn_wmma<<<B * (N / 16), 32, 0, stream>>>(adj, wh2_h, s2a, s2b, m2, s2, out);
}
